// ConvLocalBlock_58102317580330
// MI455X (gfx1250) — matrix operand order repaired and checked by reading, not compiled, not run on hardware
//
#include <hip/hip_runtime.h>
#include <hip/hip_bf16.h>
#include <stdint.h>

typedef __attribute__((ext_vector_type(16))) __bf16 v16bf;
typedef __attribute__((ext_vector_type(8)))  float  v8f;
typedef __attribute__((ext_vector_type(4)))  int    v4i;

union ABits { unsigned int u[8]; v16bf v; };

#define ASYNC_OK (__has_builtin(__builtin_amdgcn_global_load_async_to_lds_b128) && \
                  __has_builtin(__builtin_amdgcn_s_wait_asynccnt))

__device__ __forceinline__ unsigned short f2bf(float x) {
  union { float f; unsigned int u; } t; t.f = x;
  unsigned int r = t.u + 0x7FFFu + ((t.u >> 16) & 1u);
  return (unsigned short)(r >> 16);
}

#if ASYNC_OK
__device__ __forceinline__ void async_cp16(const void* g, void* s) {
  auto gp = (__attribute__((address_space(1))) v4i*)(unsigned long long)g;
  auto sp = (__attribute__((address_space(3))) v4i*)(unsigned int)(unsigned long long)s;
  __builtin_amdgcn_global_load_async_to_lds_b128(gp, sp, 0, 0);
}
__device__ __forceinline__ void async_wait() {
  __builtin_amdgcn_s_wait_asynccnt(0);
}
#else
__device__ __forceinline__ void async_cp16(const void* g, void* s) {
  *(uint4*)s = *(const uint4*)g;
}
__device__ __forceinline__ void async_wait() {}
#endif

// ---------------------------------------------------------------------------
// Kernel 0: conv_w (7,64,128 f32) -> bf16, transposed per tap to [7][128][64]
// (f-major, cin-minor) so per-tap tiles async-copy straight into the
// K-major LDS layout used by the B-fragment b128 loads.
// ---------------------------------------------------------------------------
__global__ void k_cvt(const float* __restrict__ w, unsigned short* __restrict__ o, int n) {
  int i = blockIdx.x * blockDim.x + threadIdx.x;
  if (i < n) {
    int kk = i >> 13;          // tap
    int r  = i & 8191;
    int f  = r >> 6;           // output channel
    int c  = r & 63;           // input channel
    o[i] = f2bf(w[((size_t)kk * 64 + c) * 128 + f]);
  }
}

// ---------------------------------------------------------------------------
// Kernel 1: Conv1D(same, K=7) + bias + BN1 + ReLU -> y bf16 (B,512,128)
// Workgroup: one (batch b, 64-position chunk). GEMM M=64 x N=128 x K=448,
// chunked per tap. Weight tile [128 f][64 c] bf16 async-staged to LDS.
// ---------------------------------------------------------------------------
__global__ __launch_bounds__(128)
void k_conv(const float* __restrict__ x, const unsigned short* __restrict__ wbf,
            const float* __restrict__ cb,
            const float* __restrict__ g1, const float* __restrict__ b1,
            const float* __restrict__ m1, const float* __restrict__ v1,
            unsigned short* __restrict__ yout) {
  __shared__ __align__(16) unsigned short sX[70 * 64];   // rows l0-3 .. l0+66
  __shared__ __align__(16) unsigned short sW[128 * 64];  // [f][c], K-major

  const int tid  = threadIdx.x;
  const int lane = tid & 31;
  const int wave = tid >> 5;
  const int b    = blockIdx.y;
  const int l0   = blockIdx.x * 64;

  for (int e = tid; e < 70 * 64; e += 128) {
    int r = e >> 6, c = e & 63;
    int p = l0 - 3 + r;
    float val = (p >= 0 && p < 512) ? x[((size_t)b * 512 + p) * 64 + c] : 0.0f;
    sX[e] = f2bf(val);
  }

  v8f acc[8];
#pragma unroll
  for (int i = 0; i < 8; ++i)
#pragma unroll
    for (int j = 0; j < 8; ++j) acc[i][j] = 0.0f;

  const int m0   = wave * 16;
  const int mrow = lane & 15;
  const int kA   = (lane >> 4) * 8;   // A-frag K base / 2 (u32 units)
  const int kB2  = (lane >> 4) * 8;   // B-frag K base / 2 (u32 units)
  const int ncol = lane & 15;

  const unsigned int* sWu = (const unsigned int*)sW;

  for (int kc = 0; kc < 7; ++kc) {          // one tap per chunk, K_local = 64
    __syncthreads();
    {   // async copy one tap's 128x64 bf16 tile (contiguous) into LDS
      const unsigned short* gsrc = wbf + kc * 8192;
      for (int i = tid; i < 1024; i += 128)
        async_cp16(gsrc + i * 8, &sW[i * 8]);
      async_wait();
    }
    __syncthreads();

#pragma unroll
    for (int ks = 0; ks < 2; ++ks) {        // two 16x16x32 K-steps per chunk
      ABits a;
#pragma unroll
      for (int v = 0; v < 8; ++v) {
        int k0 = ks * 32 + 2 * kA + 2 * v;
        int row = m0 + mrow + kc;
        a.u[v] = *(const unsigned int*)&sX[row * 64 + k0];
      }
      ABits bbf[8];
#pragma unroll
      for (int nt = 0; nt < 8; ++nt) {
        int n = nt * 16 + ncol;
#pragma unroll
        for (int v = 0; v < 8; ++v)
          bbf[nt].u[v] = sWu[n * 32 + ks * 16 + kB2 + v];   // 2x ds_load_b128
      }
#pragma unroll
      for (int nt = 0; nt < 8; ++nt)
        acc[nt] = __builtin_amdgcn_wmma_f32_16x16x32_bf16(
            false, a.v, false, bbf[nt].v, (short)0, acc[nt], false, false);
    }
  }

  // fused bias + BN1 + ReLU epilogue, store bf16
#pragma unroll
  for (int nt = 0; nt < 8; ++nt) {
    int f = nt * 16 + ncol;
    float bias = cb[f];
    float sc   = g1[f] * rsqrtf(v1[f] + 1e-3f);
    float mm   = m1[f], bt = b1[f];
#pragma unroll
    for (int v = 0; v < 8; ++v) {
      int Mv  = v + ((lane >> 4) << 3);
      int pos = l0 + m0 + Mv;
      float t = sc * (acc[nt][v] + bias - mm) + bt;
      t = fmaxf(t, 0.0f);
      yout[((size_t)b * 512 + pos) * 128 + f] = f2bf(t);
    }
  }
}

// ---------------------------------------------------------------------------
// Kernel 2: LocallyConnected1D + bias + BN2 + ReLU -> out f32 (64,506,128)
// Workgroup: one position l. GEMM M=64(batch) x N=128 x K=896, 14 chunks.
// y-patch rows contiguous -> async b128 to LDS; local_w chunk converted
// f32->bf16 into K-major LDS [f][k]; next chunk prefetched.
// ---------------------------------------------------------------------------
__global__ __launch_bounds__(128)
void k_local(const unsigned short* __restrict__ y, const float* __restrict__ lw,
             const float* __restrict__ lb,
             const float* __restrict__ g2, const float* __restrict__ b2,
             const float* __restrict__ m2, const float* __restrict__ v2,
             float* __restrict__ out) {
  __shared__ __align__(16) unsigned short sA[64 * 64];   // [batch][k]
  __shared__ __align__(16) unsigned short sB[128 * 64];  // [f][k], K-major

  const int tid  = threadIdx.x;
  const int lane = tid & 31;
  const int wave = tid >> 5;
  const int l    = blockIdx.x;

  const int m0   = wave * 16;
  const int mrow = lane & 15;
  const int kA   = (lane >> 4) * 8;
  const int kB2  = (lane >> 4) * 8;
  const int ncol = lane & 15;

  v8f acc[8];
#pragma unroll
  for (int i = 0; i < 8; ++i)
#pragma unroll
    for (int j = 0; j < 8; ++j) acc[i][j] = 0.0f;

  const float* lwL = lw + (size_t)l * 896 * 128;
  const unsigned int* sBu = (const unsigned int*)sB;

  for (int kc = 0; kc < 14; ++kc) {
    __syncthreads();
    // sA: async copy 64 rows x 64 bf16 (each row contiguous in y)
    for (int i = tid; i < 512; i += 128) {
      int brow = i >> 3;
      int klb  = (i & 7) * 8;
      async_cp16(y + ((size_t)brow * 512 + l) * 128 + kc * 64 + klb,
                 &sA[brow * 64 + klb]);
    }
    // sB: convert 64x128 f32 weight chunk to bf16, transposed to [f][k]
    const float* gw = lwL + kc * 64 * 128;
    for (int e = tid; e < 8192; e += 128) {
      int kl = e >> 7, f = e & 127;          // coalesced global read
      sB[f * 64 + kl] = f2bf(gw[e]);
    }
    if (kc + 1 < 14) __builtin_prefetch(gw + 8192 + tid * 16, 0, 0);
    async_wait();
    __syncthreads();

#pragma unroll
    for (int ks = 0; ks < 2; ++ks) {
      ABits a;
#pragma unroll
      for (int v = 0; v < 8; ++v) {
        int k0 = ks * 32 + 2 * kA + 2 * v;
        a.u[v] = *(const unsigned int*)&sA[(m0 + mrow) * 64 + k0];
      }
      ABits bbf[8];
#pragma unroll
      for (int nt = 0; nt < 8; ++nt) {
        int n = nt * 16 + ncol;
#pragma unroll
        for (int v = 0; v < 8; ++v)
          bbf[nt].u[v] = sBu[n * 32 + ks * 16 + kB2 + v];   // 2x ds_load_b128
      }
#pragma unroll
      for (int nt = 0; nt < 8; ++nt)
        acc[nt] = __builtin_amdgcn_wmma_f32_16x16x32_bf16(
            false, a.v, false, bbf[nt].v, (short)0, acc[nt], false, false);
    }
  }

  // fused local bias + BN2 + ReLU, store f32
#pragma unroll
  for (int nt = 0; nt < 8; ++nt) {
    int f = nt * 16 + ncol;
    float bias = lb[(size_t)l * 128 + f];
    float sc   = g2[f] * rsqrtf(v2[f] + 1e-3f);
    float mm   = m2[f], bt = b2[f];
#pragma unroll
    for (int v = 0; v < 8; ++v) {
      int bg  = m0 + v + ((lane >> 4) << 3);
      float t = sc * (acc[nt][v] + bias - mm) + bt;
      out[((size_t)bg * 506 + l) * 128 + f] = fmaxf(t, 0.0f);
    }
  }
}

// ---------------------------------------------------------------------------
extern "C" void kernel_launch(void* const* d_in, const int* in_sizes, int n_in,
                              void* d_out, int out_size, void* d_ws, size_t ws_size,
                              hipStream_t stream) {
  const float* x       = (const float*)d_in[0];
  const float* conv_w  = (const float*)d_in[1];
  const float* conv_b  = (const float*)d_in[2];
  const float* bn1_g   = (const float*)d_in[3];
  const float* bn1_b   = (const float*)d_in[4];
  const float* bn1_m   = (const float*)d_in[5];
  const float* bn1_v   = (const float*)d_in[6];
  const float* local_w = (const float*)d_in[7];
  const float* local_b = (const float*)d_in[8];
  const float* bn2_g   = (const float*)d_in[9];
  const float* bn2_b   = (const float*)d_in[10];
  const float* bn2_m   = (const float*)d_in[11];
  const float* bn2_v   = (const float*)d_in[12];
  float* out = (float*)d_out;

  unsigned short* wbf = (unsigned short*)d_ws;                       // 114,688 B
  unsigned short* ybf = (unsigned short*)((char*)d_ws + 131072);     // 8,388,608 B

  // 1) conv weights -> bf16, transposed to [tap][f][cin]
  k_cvt<<<224, 256, 0, stream>>>(conv_w, wbf, 7 * 64 * 128);
  // 2) conv + BN1 + ReLU (bf16 WMMA, f32 accum)
  k_conv<<<dim3(8, 64), 128, 0, stream>>>(x, wbf, conv_b,
                                          bn1_g, bn1_b, bn1_m, bn1_v, ybf);
  // 3) locally-connected + BN2 + ReLU (bf16 WMMA, f32 accum)
  k_local<<<506, 128, 0, stream>>>(ybf, local_w, local_b,
                                   bn2_g, bn2_b, bn2_m, bn2_v, out);
}
